// MoE_25409026523797
// MI455X (gfx1250) — compile-verified, matrix operand order checked by reading, not run on hardware
//
#include <hip/hip_runtime.h>
#include <hip/hip_bf16.h>
#include <math.h>

typedef __attribute__((ext_vector_type(16))) __bf16 v16bf;
typedef __attribute__((ext_vector_type(8)))  __bf16 v8bf;
typedef __attribute__((ext_vector_type(4)))  __bf16 v4bf;
typedef __attribute__((ext_vector_type(8)))  float  v8f;

#define D_DIM   1024
#define ED_DIM  2048
#define HSTRIDE (128 + 8)     // bf16 elems; 272 B rows: 16B-aligned, bank-staggered
#define XSTRIDE (D_DIM + 8)   // fallback path only

// ---- fragment builders (ISA 16-bit A/B layout, wave32) ---------------------
// lane<16:  elems 0..7 = K d+0..7,   8..15 = K d+16..23   (koff = 0)
// lane>=16: elems 0..7 = K d+8..15,  8..15 = K d+24..31   (koff = 8)
__device__ __forceinline__ v16bf frag_bf16(const __bf16* p, int d, int koff) {
  v8bf lo = *reinterpret_cast<const v8bf*>(p + d + koff);        // *_load_b128
  v8bf hi = *reinterpret_cast<const v8bf*>(p + d + 16 + koff);   // *_load_b128
  v16bf r;
#pragma unroll
  for (int i = 0; i < 8; ++i) { r[i] = lo[i]; r[8 + i] = hi[i]; }
  return r;
}

__device__ __forceinline__ v16bf frag_f32(const float* __restrict__ p, int d, int koff) {
  const float4* p0 = reinterpret_cast<const float4*>(p + d + koff);
  const float4* p1 = reinterpret_cast<const float4*>(p + d + 16 + koff);
  float4 a0 = p0[0], a1 = p0[1], b0 = p1[0], b1 = p1[1];
  v16bf r;
  r[0]=(__bf16)a0.x; r[1]=(__bf16)a0.y; r[2]=(__bf16)a0.z; r[3]=(__bf16)a0.w;
  r[4]=(__bf16)a1.x; r[5]=(__bf16)a1.y; r[6]=(__bf16)a1.z; r[7]=(__bf16)a1.w;
  r[8]=(__bf16)b0.x; r[9]=(__bf16)b0.y; r[10]=(__bf16)b0.z; r[11]=(__bf16)b0.w;
  r[12]=(__bf16)b1.x; r[13]=(__bf16)b1.y; r[14]=(__bf16)b1.z; r[15]=(__bf16)b1.w;
  return r;
}

// ---- weight pre-conversion: fp32 -> bf16, one pass -------------------------
__global__ __launch_bounds__(256) void cvt_bf16(const float* __restrict__ src,
                                                __bf16* __restrict__ dst, int n4) {
  int idx = blockIdx.x * 256 + threadIdx.x;
  if (idx < n4) {
    float4 v = reinterpret_cast<const float4*>(src)[idx];
    v4bf o;
    o[0] = (__bf16)v.x; o[1] = (__bf16)v.y; o[2] = (__bf16)v.z; o[3] = (__bf16)v.w;
    *reinterpret_cast<v4bf*>(dst + (size_t)idx * 4) = o;
  }
}

// ---- gating: s[t] = top2mass/(top2mass+1e-9); optionally emit bf16 x -------
__global__ __launch_bounds__(256) void moe_gate(const float* __restrict__ x,
                                                const float* __restrict__ gate_w,
                                                float* __restrict__ scale,
                                                __bf16* __restrict__ x_bf) {
  __shared__ float xrow[8][D_DIM];
  const int lane = threadIdx.x & 31;
  const int wave = threadIdx.x >> 5;
  const int t = blockIdx.x * 8 + wave;

  // cooperative, coalesced read of the token row; fused bf16 emission
  const float4* x4 = reinterpret_cast<const float4*>(x + (size_t)t * D_DIM);
  float4* l4 = reinterpret_cast<float4*>(&xrow[wave][0]);
#pragma unroll
  for (int i = 0; i < 8; ++i) {
    int d4 = i * 32 + lane;
    float4 v = x4[d4];
    l4[d4] = v;
    if (x_bf) {
      v4bf o;
      o[0] = (__bf16)v.x; o[1] = (__bf16)v.y; o[2] = (__bf16)v.z; o[3] = (__bf16)v.w;
      *reinterpret_cast<v4bf*>(x_bf + (size_t)t * D_DIM + d4 * 4) = o;
    }
  }
  // same-wave LDS RAW: DS ops are in-order per wave (compiler inserts s_wait_dscnt)
  const float4* g4a = reinterpret_cast<const float4*>(gate_w + (size_t)lane * D_DIM);
  const float4* g4b = reinterpret_cast<const float4*>(gate_w + (size_t)(lane + 32) * D_DIM);
  float acc0 = 0.f, acc1 = 0.f;
#pragma unroll 4
  for (int d4 = 0; d4 < D_DIM / 4; ++d4) {
    float4 xv = l4[d4];
    float4 ga = g4a[d4];
    float4 gb = g4b[d4];
    acc0 += xv.x * ga.x + xv.y * ga.y + xv.z * ga.z + xv.w * ga.w;
    acc1 += xv.x * gb.x + xv.y * gb.y + xv.z * gb.z + xv.w * gb.w;
  }
  float m = fmaxf(acc0, acc1);
#pragma unroll
  for (int off = 16; off > 0; off >>= 1) m = fmaxf(m, __shfl_xor(m, off, 32));
  float p0 = __expf(acc0 - m), p1 = __expf(acc1 - m);
  float s = p0 + p1;
#pragma unroll
  for (int off = 16; off > 0; off >>= 1) s += __shfl_xor(s, off, 32);
  float hi = fmaxf(p0, p1), lo = fminf(p0, p1);
#pragma unroll
  for (int off = 16; off > 0; off >>= 1) {
    float ohi = __shfl_xor(hi, off, 32);
    float olo = __shfl_xor(lo, off, 32);
    float nhi = fmaxf(hi, ohi);
    float nlo = fmaxf(fminf(hi, ohi), fmaxf(lo, olo));
    hi = nhi; lo = nlo;
  }
  if (lane == 0) {
    float mass = (hi + lo) / s;
    scale[t] = mass / (mass + 1e-9f);
  }
}

// ---- main path: fused FFN, all-bf16 operands, M_TILE = 32 ------------------
__global__ __launch_bounds__(256) void moe_ffn_bf16(const __bf16* __restrict__ x_bf,
                                                    const __bf16* __restrict__ up_bf,
                                                    const __bf16* __restrict__ dn_bf,
                                                    const float* __restrict__ scale,
                                                    float* __restrict__ out) {
  __shared__ __bf16 h_lds[32 * HSTRIDE];   // 32 x 128 bf16 (+pad)  ~8.7 KB

  const int lane = threadIdx.x & 31;
  const int wave = threadIdx.x >> 5;       // owns out cols [wave*128, +128)
  const int t0 = blockIdx.x * 32;
  const int r16 = lane & 15;
  const int koff = (lane >> 4) ? 8 : 0;
  const int tok_base = (lane >> 4) ? 8 : 0;

  const __bf16* arow0 = x_bf + (size_t)(t0 + r16) * D_DIM;
  const __bf16* arow1 = x_bf + (size_t)(t0 + 16 + r16) * D_DIM;

  v8f acc[2][8];
  {
    v8f z = {};
#pragma unroll
    for (int mt = 0; mt < 2; ++mt)
#pragma unroll
      for (int i = 0; i < 8; ++i) acc[mt][i] = z;
  }

  for (int ec = 0; ec < ED_DIM / 128; ++ec) {
    // phase 1: this wave computes h[0..32, wave*16 .. +16) of the 128-wide chunk
    const int upcol = ec * 128 + wave * 16 + r16;
    const __bf16* uprow = up_bf + (size_t)upcol * D_DIM;
    if (ec + 1 < ED_DIM / 128)
      __builtin_prefetch(up_bf + (size_t)(upcol + 128) * D_DIM, 0, 0);
    v8f h0 = {}, h1 = {};
#pragma unroll 8
    for (int kk = 0; kk < D_DIM / 32; ++kk) {
      v16bf b  = frag_bf16(uprow, kk * 32, koff);   // shared by both M tiles
      v16bf a0 = frag_bf16(arow0, kk * 32, koff);
      v16bf a1 = frag_bf16(arow1, kk * 32, koff);
      h0 = __builtin_amdgcn_wmma_f32_16x16x32_bf16(false, a0, false, b, (short)0, h0, false, false);
      h1 = __builtin_amdgcn_wmma_f32_16x16x32_bf16(false, a1, false, b, (short)0, h1, false, false);
    }
    // silu -> bf16 -> shared h chunk (C layout: vgpr r -> token r (+8 hi half))
    {
      const int hcol = wave * 16 + r16;
#pragma unroll
      for (int r = 0; r < 8; ++r) {
        float v0 = h0[r];
        h_lds[(tok_base + r) * HSTRIDE + hcol] = (__bf16)(v0 / (1.f + __expf(-v0)));
        float v1 = h1[r];
        h_lds[(16 + tok_base + r) * HSTRIDE + hcol] = (__bf16)(v1 / (1.f + __expf(-v1)));
      }
    }
    __syncthreads();

    // phase 2: acc[32 x 128] += h(32x128) @ down_w[cols, chunk]'
    // kk-outer nest: A frags loaded once per kk (8x fewer ds loads), B streamed
    // with immediate-offset addressing off one base pointer.
    const __bf16* hrow0 = &h_lds[r16 * HSTRIDE];
    const __bf16* hrow1 = &h_lds[(16 + r16) * HSTRIDE];
    const __bf16* dnbase = dn_bf + (size_t)(wave * 128 + r16) * ED_DIM + ec * 128;
#pragma unroll
    for (int kk = 0; kk < 4; ++kk) {
      v16bf a0 = frag_bf16(hrow0, kk * 32, koff);
      v16bf a1 = frag_bf16(hrow1, kk * 32, koff);
#pragma unroll
      for (int nt = 0; nt < 8; ++nt) {
        v16bf b = frag_bf16(dnbase + (size_t)nt * 16 * ED_DIM, kk * 32, koff);
        acc[0][nt] = __builtin_amdgcn_wmma_f32_16x16x32_bf16(false, a0, false, b, (short)0, acc[0][nt], false, false);
        acc[1][nt] = __builtin_amdgcn_wmma_f32_16x16x32_bf16(false, a1, false, b, (short)0, acc[1][nt], false, false);
      }
    }
    __syncthreads();
  }

  // epilogue: apply gate mass, store fp32 out
#pragma unroll
  for (int mt = 0; mt < 2; ++mt) {
#pragma unroll
    for (int r = 0; r < 8; ++r) {
      const int tok = t0 + mt * 16 + tok_base + r;
      const float sc = scale[tok];
      float* orow = out + (size_t)tok * D_DIM + wave * 128 + r16;
#pragma unroll
      for (int nt = 0; nt < 8; ++nt) orow[nt * 16] = acc[mt][nt][r] * sc;
    }
  }
}

// ---- fallback path (tiny d_ws): fp32 weights converted in-loop, M_TILE=16 --
__global__ __launch_bounds__(256) void moe_ffn_f32(const float* __restrict__ x,
                                                   const float* __restrict__ up_w,
                                                   const float* __restrict__ down_w,
                                                   const float* __restrict__ scale,
                                                   float* __restrict__ out) {
  __shared__ __bf16 x_lds[16 * XSTRIDE];
  __shared__ __bf16 h_lds[16 * HSTRIDE];
  const int lane = threadIdx.x & 31;
  const int wave = threadIdx.x >> 5;
  const int t0 = blockIdx.x * 16;
  {
    const float4* xs = reinterpret_cast<const float4*>(x + (size_t)t0 * D_DIM);
#pragma unroll
    for (int i = 0; i < 16; ++i) {
      int idx = i * 256 + threadIdx.x;
      int row = idx / (D_DIM / 4), col4 = idx % (D_DIM / 4);
      float4 v = xs[idx];
      __bf16* dst = &x_lds[row * XSTRIDE + col4 * 4];
      dst[0] = (__bf16)v.x; dst[1] = (__bf16)v.y; dst[2] = (__bf16)v.z; dst[3] = (__bf16)v.w;
    }
  }
  __syncthreads();
  const int r16 = lane & 15;
  const int koff = (lane >> 4) ? 8 : 0;
  const int tok_base = (lane >> 4) ? 8 : 0;
  const __bf16* a_row = &x_lds[r16 * XSTRIDE];
  v8f acc[8];
  { v8f z = {}; for (int i = 0; i < 8; ++i) acc[i] = z; }
  for (int ec = 0; ec < ED_DIM / 128; ++ec) {
    const float* uprow = up_w + (size_t)(ec * 128 + wave * 16 + r16) * D_DIM;
    v8f hc = {};
#pragma unroll 8
    for (int kk = 0; kk < D_DIM / 32; ++kk) {
      v16bf a = frag_bf16(a_row, kk * 32, koff);
      v16bf b = frag_f32(uprow, kk * 32, koff);
      hc = __builtin_amdgcn_wmma_f32_16x16x32_bf16(false, a, false, b, (short)0, hc, false, false);
    }
    const int hcol = wave * 16 + r16;
#pragma unroll
    for (int r = 0; r < 8; ++r) {
      float v = hc[r];
      h_lds[(tok_base + r) * HSTRIDE + hcol] = (__bf16)(v / (1.f + __expf(-v)));
    }
    __syncthreads();
    const __bf16* h_row = &h_lds[r16 * HSTRIDE];
#pragma unroll
    for (int kk = 0; kk < 4; ++kk) {
      v16bf a = frag_bf16(h_row, kk * 32, koff);
#pragma unroll
      for (int nt = 0; nt < 8; ++nt) {
        const float* dnrow = down_w + (size_t)(wave * 128 + nt * 16 + r16) * ED_DIM + ec * 128;
        v16bf b = frag_f32(dnrow, kk * 32, koff);
        acc[nt] = __builtin_amdgcn_wmma_f32_16x16x32_bf16(false, a, false, b, (short)0, acc[nt], false, false);
      }
    }
    __syncthreads();
  }
#pragma unroll
  for (int r = 0; r < 8; ++r) {
    const int tok = t0 + tok_base + r;
    const float sc = scale[tok];
    float* orow = out + (size_t)tok * D_DIM + wave * 128 + r16;
#pragma unroll
    for (int nt = 0; nt < 8; ++nt) orow[nt * 16] = acc[nt][r] * sc;
  }
}

extern "C" void kernel_launch(void* const* d_in, const int* in_sizes, int n_in,
                              void* d_out, int out_size, void* d_ws, size_t ws_size,
                              hipStream_t stream) {
  const float* x      = (const float*)d_in[0];
  const float* gate_w = (const float*)d_in[1];
  const float* up_w   = (const float*)d_in[2];
  const float* down_w = (const float*)d_in[3];
  float* out   = (float*)d_out;
  float* scale = (float*)d_ws;                       // first 32 KB of scratch

  const int T = in_sizes[0] / D_DIM;                 // 8192 tokens

  const size_t scale_b = 32768;
  const size_t xbf_b   = (size_t)T * D_DIM * 2;      // 16 MB
  const size_t up_b    = (size_t)ED_DIM * D_DIM * 2; // 4 MB
  const size_t dn_b    = (size_t)D_DIM * ED_DIM * 2; // 4 MB

  if (ws_size >= scale_b + xbf_b + up_b + dn_b) {
    __bf16* x_bf  = (__bf16*)((char*)d_ws + scale_b);
    __bf16* up_bf = (__bf16*)((char*)d_ws + scale_b + xbf_b);
    __bf16* dn_bf = (__bf16*)((char*)d_ws + scale_b + xbf_b + up_b);
    const int wn4 = ED_DIM * D_DIM / 4;              // float4 count per weight
    cvt_bf16<<<wn4 / 256, 256, 0, stream>>>(up_w, up_bf, wn4);
    cvt_bf16<<<wn4 / 256, 256, 0, stream>>>(down_w, dn_bf, wn4);
    moe_gate<<<T / 8, 256, 0, stream>>>(x, gate_w, scale, x_bf);
    moe_ffn_bf16<<<T / 32, 256, 0, stream>>>(x_bf, up_bf, dn_bf, scale, out);
  } else {
    moe_gate<<<T / 8, 256, 0, stream>>>(x, gate_w, scale, nullptr);
    moe_ffn_f32<<<T / 16, 256, 0, stream>>>(x, up_w, down_w, scale, out);
  }
}
